// BotRGCN_fmoe1_52518860095659
// MI455X (gfx1250) — compile-verified, matrix operand order checked by reading, not run on hardware
//
#include <hip/hip_runtime.h>
#include <hip/hip_bf16.h>
#include <cstdint>

// ---------------------------------------------------------------------------
// BotRGCN forward for MI455X (gfx1250): bf16 WMMA GEMMs + f32 atomics for
// the graph aggregation. All matrix math runs through v_wmma_f32_16x16x32_bf16.
// ---------------------------------------------------------------------------

typedef __bf16 bf16;
typedef __attribute__((ext_vector_type(8)))  __bf16 v8bf;
typedef __attribute__((ext_vector_type(16))) __bf16 v16bf;
typedef __attribute__((ext_vector_type(8)))  float  v8f;

static constexpr int Nn = 50000;   // nodes
static constexpr int Ed = 400000;  // edges
static constexpr int Dm = 256;     // embedding dim

__device__ __forceinline__ bf16 to_bf16(float f) {
  uint32_t u = __builtin_bit_cast(uint32_t, f);
  u += 0x7FFFu + ((u >> 16) & 1u);          // round-to-nearest-even
  uint16_t h = (uint16_t)(u >> 16);
  return __builtin_bit_cast(bf16, h);
}

// ---------------------------------------------------------------------------
// Generic bf16 GEMM:  out = act( A(MxK) * Bt^T + bias [+ Cacc] )
// Bt is B transposed, row-major [Nc][K], bf16.
// Block = 128 threads = 4 waves. Block tile: 16 rows x 256 cols.
// Each wave: 16 rows x 64 cols = 4 WMMA C-fragments; K stepped by 32.
// act: 0=none, 1=SELU, 2=LeakyReLU(0.01)
// ---------------------------------------------------------------------------
__global__ __launch_bounds__(128)
void gemm_bf16_wmma(const bf16* __restrict__ A, int lda,
                    const bf16* __restrict__ Bt, int K,
                    const float* __restrict__ bias,
                    const float* __restrict__ Cacc, int ldc,
                    float* __restrict__ outF, bf16* __restrict__ outB, int ldo,
                    int act) {
  const int lane = threadIdx.x & 31;
  const int wave = threadIdx.x >> 5;              // 0..3
  const int row0 = blockIdx.y * 16;
  const int col0 = blockIdx.x * 256 + wave * 64;
  const int half = lane >> 4;                     // K-half selector
  const int l16  = lane & 15;

  v8f c0 = {0.f,0.f,0.f,0.f,0.f,0.f,0.f,0.f};
  v8f c[4] = {c0, c0, c0, c0};

  // A fragment base: row = row0 + (lane&15); elements 0..7 = K-halfbase..+7
  const bf16* aRow = A + (size_t)(row0 + l16) * lda + half * 8;

  for (int k = 0; k < K; k += 32) {
    v8bf alo = *(const v8bf*)(aRow + k);
    v8bf ahi = *(const v8bf*)(aRow + k + 16);
    v16bf a = __builtin_shufflevector(alo, ahi,
        0,1,2,3,4,5,6,7,8,9,10,11,12,13,14,15);
#pragma unroll
    for (int t = 0; t < 4; ++t) {
      const bf16* bRow = Bt + (size_t)(col0 + t * 16 + l16) * K + half * 8;
      v8bf blo = *(const v8bf*)(bRow + k);
      v8bf bhi = *(const v8bf*)(bRow + k + 16);
      v16bf b = __builtin_shufflevector(blo, bhi,
          0,1,2,3,4,5,6,7,8,9,10,11,12,13,14,15);
      c[t] = __builtin_amdgcn_wmma_f32_16x16x32_bf16(
          /*neg_a=*/false, a, /*neg_b=*/false, b,
          /*c_mod=*/(short)0, c[t], /*reuse_a=*/false, /*reuse_b=*/false);
    }
  }

  const float selu_scale = 1.0507009873554805f;
  const float selu_alpha = 1.6732632423543772f;
#pragma unroll
  for (int t = 0; t < 4; ++t) {
    int col = col0 + t * 16 + l16;
    float bv = bias ? bias[col] : 0.0f;
#pragma unroll
    for (int v = 0; v < 8; ++v) {
      int row = row0 + v + half * 8;              // C/D layout per ISA
      float val = c[t][v] + bv;
      if (Cacc) val += Cacc[(size_t)row * ldc + col];
      if (act == 1) {
        val = (val > 0.0f) ? selu_scale * val
                           : selu_scale * selu_alpha * (__expf(val) - 1.0f);
      } else if (act == 2) {
        val = (val >= 0.0f) ? val : 0.01f * val;
      }
      if (outF) outF[(size_t)row * ldo + col] = val;
      if (outB) outB[(size_t)row * ldo + col] = to_bf16(val);
    }
  }
}

// ---------------------------------------------------------------------------
// Weight prep: transpose + bf16-convert all weight matrices once per call.
// ---------------------------------------------------------------------------
__global__ void prep_weights(const float* __restrict__ W_in,
                             const float* __restrict__ W_rel,
                             const float* __restrict__ W_root,
                             const float* __restrict__ We1,
                             const float* __restrict__ We2,
                             bf16* WinT, bf16* WrelT, bf16* WrootT,
                             bf16* We1T, bf16* We2T) {
  const int total = 65536 + 65536 + 131072 + 131072 + 131072;
  for (int idx = blockIdx.x * blockDim.x + threadIdx.x; idx < total;
       idx += gridDim.x * blockDim.x) {
    int i = idx;
    if (i < 65536) { int n = i >> 8, k = i & 255;
      WinT[n * 256 + k] = to_bf16(W_in[k * 256 + n]); continue; }
    i -= 65536;
    if (i < 65536) { int n = i >> 8, k = i & 255;
      WrootT[n * 256 + k] = to_bf16(W_root[k * 256 + n]); continue; }
    i -= 65536;
    if (i < 131072) { int n = i >> 9, k = i & 511;   // stacked relations: K=512
      float v = (k < 256) ? W_rel[k * 256 + n]
                          : W_rel[65536 + (k - 256) * 256 + n];
      WrelT[n * 512 + k] = to_bf16(v); continue; }
    i -= 131072;
    if (i < 131072) { int n = i >> 8, k = i & 255;   // stacked experts: Nc=512
      int e = n >> 8, h = n & 255;
      We1T[n * 256 + k] = to_bf16(We1[e * 65536 + k * 256 + h]); continue; }
    i -= 131072;
    { int e = i >> 16, j = i & 65535; int n = j >> 8, k = j & 255;
      We2T[e * 65536 + n * 256 + k] = to_bf16(We2[e * 65536 + k * 256 + n]); }
  }
}

// Concatenate (des|tweet|num_prop|cat_prop) into bf16 staging [N,256].
__global__ void concat_kernel(const float* __restrict__ des,
                              const float* __restrict__ tw,
                              const float* __restrict__ np,
                              const float* __restrict__ cp,
                              bf16* __restrict__ catB) {
  int idx = blockIdx.x * 256 + threadIdx.x;
  if (idx >= Nn * Dm) return;
  int n = idx >> 8, d = idx & 255;
  float v;
  if (d < 128)      v = des[n * 128 + d];
  else if (d < 192) v = tw[n * 64 + d - 128];
  else if (d < 224) v = np[n * 32 + d - 192];
  else              v = cp[n * 32 + d - 224];
  catB[idx] = to_bf16(v);
}

__global__ void zero_kernel(float* __restrict__ p, int n) {
  int i = blockIdx.x * 256 + threadIdx.x;
  if (i < n) p[i] = 0.0f;
}

// Scatter raw features per (dst, relation) segment (mean is linear, so we can
// aggregate BEFORE the relation GEMM). One wave per edge, 8 floats per lane.
__global__ void edge_scatter(const float* __restrict__ x,
                             const int* __restrict__ ei,
                             const int* __restrict__ et,
                             float* __restrict__ sums,
                             float* __restrict__ cnt) {
  int e = blockIdx.x * 8 + (threadIdx.x >> 5);
  if (e >= Ed) return;
  int lane = threadIdx.x & 31;
  int src = ei[e];
  int dst = ei[Ed + e];
  int r   = et[e];
  size_t seg = (size_t)dst * 2 + r;
  const float* xs = x + (size_t)src * Dm;
  float* s = sums + seg * Dm;
#pragma unroll
  for (int i = 0; i < 8; ++i)
    atomicAdd(&s[lane + i * 32], xs[lane + i * 32]);
  if (lane == 0) atomicAdd(&cnt[seg], 1.0f);
}

// mean = sums / max(cnt,1), converted to bf16 as the [N,512] stacked A matrix.
__global__ void mean_kernel(const float* __restrict__ sums,
                            const float* __restrict__ cnt,
                            bf16* __restrict__ meanB) {
  int idx = blockIdx.x * 256 + threadIdx.x;
  if (idx >= Nn * 512) return;
  int seg = idx >> 8;
  float c = fmaxf(cnt[seg], 1.0f);
  meanB[idx] = to_bf16(sums[idx] / c);
}

// Top-1 gate: one wave per node, cross-lane reduce two 256-dot-products.
__global__ void gate_kernel(const float* __restrict__ x,
                            const float* __restrict__ wg,
                            int* __restrict__ gsel) {
  int n = blockIdx.x * 8 + (threadIdx.x >> 5);
  if (n >= Nn) return;
  int lane = threadIdx.x & 31;
  const float* xr = x + (size_t)n * Dm;
  float s0 = 0.0f, s1 = 0.0f;
  for (int i = lane; i < Dm; i += 32) {
    float v = xr[i];
    s0 += v * wg[i * 2 + 0];
    s1 += v * wg[i * 2 + 1];
  }
#pragma unroll
  for (int m = 16; m >= 1; m >>= 1) {
    s0 += __shfl_xor(s0, m, 32);
    s1 += __shfl_xor(s1, m, 32);
  }
  if (lane == 0) gsel[n] = (s1 > s0) ? 1 : 0;   // tie -> lower index (top_k)
}

// y[n] = ye[selected_expert][n] (gate weight = 1.0 for top-1 softmax).
__global__ void select_kernel(const float* __restrict__ ye,
                              const int* __restrict__ gsel,
                              float* __restrict__ out) {
  int idx = blockIdx.x * 256 + threadIdx.x;
  if (idx >= Nn * Dm) return;
  int n = idx >> 8;
  out[idx] = ye[(size_t)gsel[n] * Nn * Dm + idx];
}

// ---------------------------------------------------------------------------
extern "C" void kernel_launch(void* const* d_in, const int* in_sizes, int n_in,
                              void* d_out, int out_size, void* d_ws, size_t ws_size,
                              hipStream_t stream) {
  (void)in_sizes; (void)n_in; (void)out_size; (void)ws_size;
  const float* des    = (const float*)d_in[0];
  const float* tweet  = (const float*)d_in[1];
  const float* nprop  = (const float*)d_in[2];
  const float* cprop  = (const float*)d_in[3];
  const int*   eidx2  = (const int*)d_in[4];    // (2, E)
  const int*   etype  = (const int*)d_in[5];
  const float* W_in   = (const float*)d_in[6];
  const float* b_in   = (const float*)d_in[7];
  const float* W_rel  = (const float*)d_in[8];
  const float* W_root = (const float*)d_in[9];
  const float* b_rg   = (const float*)d_in[10];
  const float* w_gate = (const float*)d_in[11];
  const float* We1    = (const float*)d_in[12];
  const float* be1    = (const float*)d_in[13]; // (2,256) flat == stacked bias
  const float* We2    = (const float*)d_in[14];
  const float* be2    = (const float*)d_in[15];
  float* out = (float*)d_out;

  // ---- workspace carve-up (256B aligned) ----
  char* base = (char*)d_ws;
  size_t off = 0;
  auto take = [&](size_t bytes) -> char* {
    char* p = base + off;
    off = (off + bytes + 255) & ~(size_t)255;
    return p;
  };
  float* xF[2]; bf16* xB[2];
  xF[0]        = (float*)take((size_t)Nn * Dm * 4);
  xF[1]        = (float*)take((size_t)Nn * Dm * 4);
  xB[0]        = (bf16*) take((size_t)Nn * Dm * 2);
  xB[1]        = (bf16*) take((size_t)Nn * Dm * 2);
  bf16*  meanB = (bf16*) take((size_t)Nn * 512 * 2); // also concat & h1 staging
  float* sums  = (float*)take((size_t)Nn * 512 * 4); // also ye0|ye1
  float* cnt   = (float*)take((size_t)Nn * 2 * 4);
  int*   gsel  = (int*)  take((size_t)Nn * 4);
  bf16*  WinT  = (bf16*) take(65536 * 2);
  bf16*  WrootT= (bf16*) take(65536 * 2);
  bf16*  WrelT = (bf16*) take(131072 * 2);
  bf16*  We1T  = (bf16*) take(131072 * 2);
  bf16*  We2T  = (bf16*) take(131072 * 2);

  const dim3 blkG(128);
  const dim3 g256(1, Nn / 16);  // Nc = 256
  const dim3 g512(2, Nn / 16);  // Nc = 512

  // 0) weight prep (bf16 transposes)
  prep_weights<<<512, 256, 0, stream>>>(W_in, W_rel, W_root, We1, We2,
                                        WinT, WrelT, WrootT, We1T, We2T);
  // 1) concat -> bf16 staging (reuse meanB)
  concat_kernel<<<(Nn * Dm + 255) / 256, 256, 0, stream>>>(des, tweet, nprop,
                                                           cprop, meanB);
  // 2) x = SELU(cat @ W_in + b_in), write f32 + bf16
  gemm_bf16_wmma<<<g256, blkG, 0, stream>>>(meanB, 256, WinT, 256, b_in,
                                            nullptr, 0, xF[0], xB[0], 256, 1);
  // 3) two RGCN layers
  int cur = 0;
  for (int layer = 0; layer < 2; ++layer) {
    zero_kernel<<<(Nn * 512 + 255) / 256, 256, 0, stream>>>(sums, Nn * 512);
    zero_kernel<<<(Nn * 2 + 255) / 256, 256, 0, stream>>>(cnt, Nn * 2);
    edge_scatter<<<Ed / 8, 256, 0, stream>>>(xF[cur], eidx2, etype, sums, cnt);
    mean_kernel<<<(Nn * 512 + 255) / 256, 256, 0, stream>>>(sums, cnt, meanB);
    int nxt = cur ^ 1;
    // agg = [mean_r0 | mean_r1] @ [W_rel0; W_rel1]   (K = 512)
    gemm_bf16_wmma<<<g256, blkG, 0, stream>>>(meanB, 512, WrelT, 512, nullptr,
                                              nullptr, 0, xF[nxt], nullptr, 256, 0);
    // x_new = x @ W_root + b + agg
    gemm_bf16_wmma<<<g256, blkG, 0, stream>>>(xB[cur], 256, WrootT, 256, b_rg,
                                              xF[nxt], 256, xF[nxt], xB[nxt], 256, 0);
    cur = nxt;
  }
  // 4) MoE: top-1 gate
  gate_kernel<<<(Nn + 7) / 8, 256, 0, stream>>>(xF[cur], w_gate, gsel);
  // h1 = LeakyReLU(x @ [We1_0 | We1_1] + be1)  -> bf16 [N,512]
  gemm_bf16_wmma<<<g512, blkG, 0, stream>>>(xB[cur], 256, We1T, 256, be1,
                                            nullptr, 0, nullptr, meanB, 512, 2);
  // ye_e = h1_e @ We2_e + be2_e  -> f32, packed into `sums`
  for (int e = 0; e < 2; ++e) {
    gemm_bf16_wmma<<<g256, blkG, 0, stream>>>(meanB + e * 256, 512,
                                              We2T + e * 65536, 256,
                                              be2 + e * 256, nullptr, 0,
                                              sums + (size_t)e * Nn * Dm,
                                              nullptr, 256, 0);
  }
  // 5) gather selected expert output
  select_kernel<<<(Nn * Dm + 255) / 256, 256, 0, stream>>>(sums, gsel, out);
}
